// MultiLayerCoarsening_37726992728634
// MI455X (gfx1250) — matrix operand order repaired and checked by reading, not compiled, not run on hardware
//
#include <hip/hip_runtime.h>
#include <math.h>

// Problem constants (from the reference): B=8, N=2048, F=256, H=128, K=1025, 2 layers.
static constexpr int Bb = 8;
static constexpr int Nn = 2048;
static constexpr int Ff = 256;
static constexpr int Hh = 128;
static constexpr int Ktop = Nn / 2 + 1;   // 1025

typedef __attribute__((ext_vector_type(16))) __bf16 v16bf;
typedef __attribute__((ext_vector_type(8)))  __bf16 v8bf;
typedef __attribute__((ext_vector_type(4)))  __bf16 v4bf;
typedef __attribute__((ext_vector_type(8)))  float  v8f;

// ---------------------------------------------------------------------------
// Batched GEMM: C[b] = A[b] @ (rowscale(S[b]) * B[b]),  f32 in/out, bf16 WMMA.
// Block: 128x128 C tile, 256 threads (8 waves), each wave 32x64 (2x4 WMMA).
// - float4 global loads staged in registers, then converted to bf16 in LDS.
// - LDS double buffered (2 x 20KB), one barrier per K-step; loads for tile t+1
//   are issued before the WMMAs of tile t so memory latency hides under math.
// - LDS rows padded to 40 bf16 (80B): 16B-aligned, conflict-free ds_load_b128.
// ---------------------------------------------------------------------------
#define GBM 128
#define GBN 128
#define GBK 32
#define GLDK 40
#define LDS_TA (GBM * GLDK)
#define LDS_TB (GBN * GLDK)

__global__ __launch_bounds__(256)
void gemm_bf16_wmma(const float* __restrict__ Ag, long long sAb,
                    const float* __restrict__ Bg, long long sBb,
                    const float* __restrict__ Sg, long long sSb, int useS,
                    float* __restrict__ Cg, long long sCb,
                    int M, int Nc, int Kd)
{
  (void)M;
  __shared__ __align__(16) __bf16 lA[2][LDS_TA];
  __shared__ __align__(16) __bf16 lB[2][LDS_TB];

  const int bz    = blockIdx.z;
  const int nBase = blockIdx.x * GBN;
  const int mBase = blockIdx.y * GBM;
  const float* A  = Ag + (long long)bz * sAb;
  const float* Bm = Bg + (long long)bz * sBb;
  const float* Sv = Sg + (long long)bz * sSb;   // dereferenced only if useS
  float*       C  = Cg + (long long)bz * sCb;

  const int tid   = threadIdx.x;
  const int lane  = tid & 31;
  const int wid   = tid >> 5;
  const int mOff  = (wid >> 1) * 32;   // 4 waves along M
  const int nOff  = (wid & 1) * 64;    // 2 waves along N
  const int lrow  = lane & 15;
  const int lhalf = lane >> 4;

  // Staging maps (32-bit in-tile offsets; one 64-bit batch-base add above).
  const int aRow0 = tid >> 3;          // A: rows aRow0 + {0,32,64,96}
  const int aCol  = (tid & 7) * 4;     //    k-offset, float4 granularity
  const int bRow0 = tid >> 5;          // B: rows bRow0 + {0,8,16,24}
  const int bCol  = (tid & 31) * 4;    //    n-offset, float4 granularity

  float4 ra[4], rb[4];
  float  sv[4];

  auto gload = [&](int k0) {
#pragma unroll
    for (int i = 0; i < 4; ++i) {
      int r = aRow0 + i * 32;
      ra[i] = *(const float4*)(A + (long long)(mBase + r) * Kd + (k0 + aCol));
    }
#pragma unroll
    for (int i = 0; i < 4; ++i) {
      int r = bRow0 + i * 8;
      rb[i] = *(const float4*)(Bm + (long long)(k0 + r) * Nc + (nBase + bCol));
      sv[i] = useS ? Sv[k0 + r] : 1.0f;
    }
  };

  auto lstore = [&](int buf) {
#pragma unroll
    for (int i = 0; i < 4; ++i) {             // A: bf16x4 -> ds_store_b64
      int r = aRow0 + i * 32;
      v4bf p;
      p[0] = (__bf16)ra[i].x; p[1] = (__bf16)ra[i].y;
      p[2] = (__bf16)ra[i].z; p[3] = (__bf16)ra[i].w;
      *(v4bf*)&lA[buf][r * GLDK + aCol] = p;
    }
#pragma unroll
    for (int i = 0; i < 4; ++i) {             // B: scaled, transposed scatter
      int r = bRow0 + i * 8;
      float f0 = rb[i].x * sv[i], f1 = rb[i].y * sv[i];
      float f2 = rb[i].z * sv[i], f3 = rb[i].w * sv[i];
      lB[buf][(bCol + 0) * GLDK + r] = (__bf16)f0;
      lB[buf][(bCol + 1) * GLDK + r] = (__bf16)f1;
      lB[buf][(bCol + 2) * GLDK + r] = (__bf16)f2;
      lB[buf][(bCol + 3) * GLDK + r] = (__bf16)f3;
    }
  };

  v8f acc[2][4];
#pragma unroll
  for (int i = 0; i < 2; ++i)
#pragma unroll
    for (int j = 0; j < 4; ++j)
#pragma unroll
      for (int r = 0; r < 8; ++r) acc[i][j][r] = 0.0f;

  const int T = Kd / GBK;
  gload(0);
  lstore(0);
  __syncthreads();

  for (int t = 0; t < T; ++t) {
    const int cur = t & 1;
    if (t + 1 < T) gload((t + 1) * GBK);   // issue next tile's global loads

    // A fragment: lane holds row M=lrow; half-K interleave {0..7,16..23}/{8..15,24..31}.
    v16bf af[2], bfr[4];
#pragma unroll
    for (int mi = 0; mi < 2; ++mi) {
      const __bf16* p = &lA[cur][(mOff + mi * 16 + lrow) * GLDK + lhalf * 8];
      v8bf lo = *(const v8bf*)p;
      v8bf hi = *(const v8bf*)(p + 16);
      af[mi] = __builtin_shufflevector(lo, hi, 0,1,2,3,4,5,6,7,8,9,10,11,12,13,14,15);
    }
    // B fragment: lane holds column N=lrow; half-K interleave {0..7,16..23}/{8..15,24..31}.
#pragma unroll
    for (int ni = 0; ni < 4; ++ni) {
      const __bf16* p = &lB[cur][(nOff + ni * 16 + lrow) * GLDK + lhalf * 8];
      v8bf lo = *(const v8bf*)p;
      v8bf hi = *(const v8bf*)(p + 16);
      bfr[ni] = __builtin_shufflevector(lo, hi, 0,1,2,3,4,5,6,7,8,9,10,11,12,13,14,15);
    }
#pragma unroll
    for (int mi = 0; mi < 2; ++mi)
#pragma unroll
      for (int ni = 0; ni < 4; ++ni)
        acc[mi][ni] = __builtin_amdgcn_wmma_f32_16x16x32_bf16(
            false, af[mi], false, bfr[ni], (short)0, acc[mi][ni], false, false);

    if (t + 1 < T) lstore(1 - cur);        // fill the other buffer
    __syncthreads();
  }

  // C/D layout: lanes 0-15 -> M = r, lanes 16-31 -> M = r+8; N = lane%16.
#pragma unroll
  for (int mi = 0; mi < 2; ++mi)
#pragma unroll
    for (int ni = 0; ni < 4; ++ni) {
      int col   = nBase + nOff + ni * 16 + lrow;
      int rbase = mBase + mOff + mi * 16 + lhalf * 8;
#pragma unroll
      for (int r = 0; r < 8; ++r)
        C[(long long)(rbase + r) * Nc + col] = acc[mi][ni][r];
    }
}

// ---------------------------------------------------------------------------
// Wave-per-row reductions (wave32).
// ---------------------------------------------------------------------------
__device__ __forceinline__ float waveSum(float s) {
#pragma unroll
  for (int o = 16; o > 0; o >>= 1) s += __shfl_xor(s, o, 32);
  return s;
}

__global__ __launch_bounds__(256)
void rowsum_diag(const float* __restrict__ A, float* rs, float* dg, int N) {
  int b = blockIdx.y;
  int row = blockIdx.x * 8 + (threadIdx.x >> 5);
  int lane = threadIdx.x & 31;
  const float* Ar = A + ((long long)b * N + row) * (long long)N;
  float s = 0.0f;
  for (int m = lane; m < N; m += 32) s += Ar[m];
  s = waveSum(s);
  if (lane == 0) { rs[(long long)b * N + row] = s; dg[(long long)b * N + row] = Ar[row]; }
}

__global__ __launch_bounds__(256)
void matvecN(const float* __restrict__ A, const float* __restrict__ v, float* outp, int N) {
  int b = blockIdx.y;
  int row = blockIdx.x * 8 + (threadIdx.x >> 5);
  int lane = threadIdx.x & 31;
  const float* Ar = A + ((long long)b * N + row) * (long long)N;
  const float* vb = v + (long long)b * N;
  float s = 0.0f;
  for (int m = lane; m < N; m += 32) s += Ar[m] * vb[m];
  s = waveSum(s);
  if (lane == 0) outp[(long long)b * N + row] = s;
}

__global__ __launch_bounds__(256)
void matvec_feat(const float* __restrict__ X, const float* __restrict__ Wt,
                 const float* __restrict__ bt, float* outp, int N, int C) {
  int b = blockIdx.y;
  int row = blockIdx.x * 8 + (threadIdx.x >> 5);
  int lane = threadIdx.x & 31;
  const float* xr = X + ((long long)b * N + row) * (long long)C;
  float s = 0.0f;
  for (int h = lane; h < C; h += 32) s += xr[h] * Wt[h];
  s = waveSum(s);
  if (lane == 0) outp[(long long)b * N + row] = s + bt[0];
}

// ---------------------------------------------------------------------------
// Elementwise / scalar kernels.
// ---------------------------------------------------------------------------
__global__ __launch_bounds__(256)
void gcn_scal(const float* rs, const float* dg, float* d, long long n) {
  long long i = (long long)blockIdx.x * 256 + threadIdx.x;
  if (i < n) d[i] = rsqrtf(fmaxf(rs[i] - dg[i] + 1.0f, 1.0f));
}

__global__ __launch_bounds__(256)
void coarsen_scal1(const float* rs, const float* v, float* d, float* rn, float* dv, long long n) {
  long long i = (long long)blockIdx.x * 256 + threadIdx.x;
  if (i < n) {
    float dd = rsqrtf(fmaxf(rs[i] + 1.0f, 1.0f));
    d[i] = dd;
    rn[i] = rs[i] > 0.0f ? 1.0f : 0.0f;
    dv[i] = dd * v[i];
  }
}

__global__ __launch_bounds__(256)
void alpha_k(const float* t, const float* rn, const float* d, const float* v, float* al, long long n) {
  long long i = (long long)blockIdx.x * 256 + threadIdx.x;
  if (i < n) {
    float ap = rn[i] * d[i] * (t[i] + d[i] * v[i]);
    al[i] = 1.0f / (1.0f + expf(-(ap * ap)));
  }
}

__global__ __launch_bounds__(256)
void cutalpha_k(const float* al, const float* cut, const float* d, float* ca, float* w,
                long long n, int N) {
  long long i = (long long)blockIdx.x * 256 + threadIdx.x;
  if (i < n) {
    int b = (int)(i / N);
    float c = fmaxf(al[i] + 1e-7f - cut[b], 0.0f);
    ca[i] = c;
    w[i] = d[i] * c;
  }
}

__global__ __launch_bounds__(256)
void u_k(const float* rn, const float* d, const float* t2, const float* ca, float* u, long long n) {
  long long i = (long long)blockIdx.x * 256 + threadIdx.x;
  if (i < n) {
    float rowS = rn[i] * d[i] * (t2[i] + d[i] * ca[i]);
    u[i] = rn[i] * d[i] / fmaxf(rowS, 1e-12f);
  }
}

__global__ __launch_bounds__(256)
void epi_gcn_relu(const float* __restrict__ Hb, const float* __restrict__ XW,
                  const float* __restrict__ dvec, const float* __restrict__ dg,
                  const float* __restrict__ bias, const float* __restrict__ mask,
                  float* outp, long long n, int N, int C) {
  long long i = (long long)blockIdx.x * 256 + threadIdx.x;
  if (i >= n) return;
  int c = (int)(i % C);
  long long bn = i / C;                 // = b*N + row
  float dd = dvec[bn];
  float val = dd * (Hb[i] + (1.0f - dg[bn]) * dd * XW[i]) + bias[c];
  outp[i] = fmaxf(val, 0.0f) * mask[bn];
}

__global__ __launch_bounds__(256)
void epi_x2(const float* __restrict__ Y, const float* __restrict__ cx,
            const float* __restrict__ u, const float* __restrict__ w,
            float* outp, long long n, int C) {
  long long i = (long long)blockIdx.x * 256 + threadIdx.x;
  if (i >= n) return;
  long long bn = i / C;
  outp[i] = w[bn] * (Y[i] + u[bn] * cx[i]);
}

__global__ __launch_bounds__(256)
void epi_Q(float* __restrict__ Q, const float* __restrict__ A, const float* __restrict__ u,
           long long n, int N) {
  long long i = (long long)blockIdx.x * 256 + threadIdx.x;
  if (i >= n) return;
  int m = (int)(i % N);
  int b = (int)(i / ((long long)N * N));
  Q[i] += A[i] * u[(long long)b * N + m];
}

__global__ __launch_bounds__(256)
void epi_A2(float* __restrict__ R, const float* __restrict__ Q,
            const float* __restrict__ u, const float* __restrict__ w, long long n, int N) {
  long long i = (long long)blockIdx.x * 256 + threadIdx.x;
  if (i >= n) return;
  int l = (int)(i % N);
  long long t = i / N;
  int m = (int)(t % N);
  int b = (int)(t / N);
  long long bm = (long long)b * N + m, bl = (long long)b * N + l;
  float r = R[i] + u[bm] * Q[i];
  R[i] = floorf(w[bm] * w[bl] * r * 1.0e4f) * 1.0e-4f;
}

// Exact K-th largest per batch: bitonic sort of 2048 floats in LDS (ascending),
// cut = sorted[N - K].
__global__ __launch_bounds__(1024)
void topk_cut(const float* __restrict__ alpha, float* cut, int N, int Kth) {
  __shared__ float buf[2048];
  int b = blockIdx.x, tid = threadIdx.x;
  for (int i = tid; i < N; i += 1024) buf[i] = alpha[(long long)b * N + i];
  __syncthreads();
  for (int k = 2; k <= N; k <<= 1) {
    for (int j = k >> 1; j > 0; j >>= 1) {
      for (int i = tid; i < N; i += 1024) {
        int ixj = i ^ j;
        if (ixj > i) {
          bool up = ((i & k) == 0);
          float a = buf[i], c = buf[ixj];
          if (up ? (a > c) : (a < c)) { buf[i] = c; buf[ixj] = a; }
        }
      }
      __syncthreads();
    }
  }
  if (tid == 0) cut[b] = buf[N - Kth];
}

// Column means.
__global__ __launch_bounds__(256)
void mean_cols(const float* __restrict__ X, float* outp, int N, int C, int outStride, int outOff) {
  __shared__ float red[256];
  int b = blockIdx.y, c = blockIdx.x, tid = threadIdx.x;
  float s = 0.0f;
  for (int n = tid; n < N; n += 256) s += X[((long long)b * N + n) * C + c];
  red[tid] = s; __syncthreads();
  for (int o = 128; o > 0; o >>= 1) { if (tid < o) red[tid] += red[tid + o]; __syncthreads(); }
  if (tid == 0) outp[(long long)b * outStride + outOff + c] = red[0] / (float)N;
}

// Final GCN epilogue fused with the column mean (avoids materializing out2).
__global__ __launch_bounds__(256)
void final_mean(const float* __restrict__ Hb, const float* __restrict__ XW,
                const float* __restrict__ dvec, const float* __restrict__ dg,
                const float* __restrict__ bias, const float* __restrict__ mask,
                float* outp, int N, int C, int outStride, int outOff) {
  __shared__ float red[256];
  int b = blockIdx.y, c = blockIdx.x, tid = threadIdx.x;
  float s = 0.0f;
  for (int n = tid; n < N; n += 256) {
    long long bn = (long long)b * N + n;
    long long i = bn * C + c;
    float dd = dvec[bn];
    float val = dd * (Hb[i] + (1.0f - dg[bn]) * dd * XW[i]) + bias[c];
    s += val * mask[bn];
  }
  red[tid] = s; __syncthreads();
  for (int o = 128; o > 0; o >>= 1) { if (tid < o) red[tid] += red[tid + o]; __syncthreads(); }
  if (tid == 0) outp[(long long)b * outStride + outOff + c] = red[0] / (float)N;
}

// ---------------------------------------------------------------------------
// Orchestration.  Workspace requirement ~490 MB:
//   3 * B*N*N (f32) big buffers + B*N*{H,F} intermediates + scalar vectors.
// ---------------------------------------------------------------------------
extern "C" void kernel_launch(void* const* d_in, const int* in_sizes, int n_in,
                              void* d_out, int out_size, void* d_ws, size_t ws_size,
                              hipStream_t stream) {
  (void)in_sizes; (void)n_in; (void)out_size; (void)ws_size;
  const float* x    = (const float*)d_in[0];
  const float* adj  = (const float*)d_in[1];
  const float* mask = (const float*)d_in[2];
  const float* W1   = (const float*)d_in[3];
  const float* b1   = (const float*)d_in[4];
  const float* Watt = (const float*)d_in[5];
  const float* batt = (const float*)d_in[6];
  const float* W2   = (const float*)d_in[7];
  const float* b2   = (const float*)d_in[8];
  float* out = (float*)d_out;
  float* ws  = (float*)d_ws;

  const long long NN  = (long long)Nn * Nn;
  const size_t BNNf = (size_t)Bb * NN;
  const size_t BNHf = (size_t)Bb * Nn * Hh;
  const size_t BNFf = (size_t)Bb * Nn * Ff;
  const size_t BNf  = (size_t)Bb * Nn;

  size_t o = 0;
  float* wsA1  = ws + o; o += BNNf;
  float* wsA2  = ws + o; o += BNNf;
  float* wsQ   = ws + o; o += BNNf;
  float* wsX1  = ws + o; o += BNHf;
  float* wsCX1 = ws + o; o += BNHf;
  float* wsCX2 = ws + o; o += BNHf;
  float* wsXW1 = ws + o; o += BNHf;
  float* wsH1  = ws + o; o += BNHf;
  float* wsY   = ws + o; o += BNHf;
  float* wsXW2 = ws + o; o += BNFf;
  float* wsH2  = ws + o; o += BNFf;
  float* wsRS  = ws + o; o += BNf;
  float* wsDG  = ws + o; o += BNf;
  float* wsD   = ws + o; o += BNf;
  float* wsRN  = ws + o; o += BNf;
  float* wsV   = ws + o; o += BNf;
  float* wsDV  = ws + o; o += BNf;
  float* wsT   = ws + o; o += BNf;
  float* wsAL  = ws + o; o += BNf;
  float* wsCA  = ws + o; o += BNf;
  float* wsW   = ws + o; o += BNf;
  float* wsU   = ws + o; o += BNf;
  float* wsT2  = ws + o; o += BNf;
  float* wsCut = ws + o; o += (size_t)Bb;

  const dim3 blk256(256);
  const dim3 rowGrid(Nn / 8, Bb);
  const int ebBN  = (int)((BNf  + 255) / 256);
  const int ebBNH = (int)((BNHf + 255) / 256);
  const int ebBNN = (int)((BNNf + 255) / 256);

  // ---- Layer 0: x1 = relu(dense_gcn(x, adj, W1, b1, mask)) ----
  rowsum_diag<<<rowGrid, blk256, 0, stream>>>(adj, wsRS, wsDG, Nn);
  gcn_scal<<<ebBN, blk256, 0, stream>>>(wsRS, wsDG, wsD, (long long)BNf);
  gemm_bf16_wmma<<<dim3(Hh / GBN, Nn / GBM, Bb), blk256, 0, stream>>>(
      x, (long long)Nn * Ff, W1, 0, nullptr, 0, 0, wsXW1, (long long)Nn * Hh, Nn, Hh, Ff);
  gemm_bf16_wmma<<<dim3(Hh / GBN, Nn / GBM, Bb), blk256, 0, stream>>>(
      adj, NN, wsXW1, (long long)Nn * Hh, wsD, Nn, 1, wsH1, (long long)Nn * Hh, Nn, Hh, Nn);
  epi_gcn_relu<<<ebBNH, blk256, 0, stream>>>(wsH1, wsXW1, wsD, wsDG, b1, mask,
                                             wsX1, (long long)BNHf, Nn, Hh);

  // ---- Two coarsening layers (S factored as u[n]*(A+I)[n,m]*w[m]) ----
  auto coarsen = [&](const float* Acur, float* Anext, const float* cxi, float* cxo) {
    rowsum_diag<<<rowGrid, blk256, 0, stream>>>(Acur, wsRS, wsDG, Nn);
    matvec_feat<<<rowGrid, blk256, 0, stream>>>(cxi, Watt, batt, wsV, Nn, Hh);
    coarsen_scal1<<<ebBN, blk256, 0, stream>>>(wsRS, wsV, wsD, wsRN, wsDV, (long long)BNf);
    matvecN<<<rowGrid, blk256, 0, stream>>>(Acur, wsDV, wsT, Nn);
    alpha_k<<<ebBN, blk256, 0, stream>>>(wsT, wsRN, wsD, wsV, wsAL, (long long)BNf);
    topk_cut<<<dim3(Bb), dim3(1024), 0, stream>>>(wsAL, wsCut, Nn, Ktop);
    cutalpha_k<<<ebBN, blk256, 0, stream>>>(wsAL, wsCut, wsD, wsCA, wsW, (long long)BNf, Nn);
    matvecN<<<rowGrid, blk256, 0, stream>>>(Acur, wsW, wsT2, Nn);
    u_k<<<ebBN, blk256, 0, stream>>>(wsRN, wsD, wsT2, wsCA, wsU, (long long)BNf);
    // x2 = diag(w) * (A @ (u .* x) + u .* x)        (A symmetric => A^T = A)
    gemm_bf16_wmma<<<dim3(Hh / GBN, Nn / GBM, Bb), blk256, 0, stream>>>(
        Acur, NN, cxi, (long long)Nn * Hh, wsU, Nn, 1, wsY, (long long)Nn * Hh, Nn, Hh, Nn);
    epi_x2<<<ebBNH, blk256, 0, stream>>>(wsY, cxi, wsU, wsW, cxo, (long long)BNHf, Hh);
    // Q = A @ (u .* A) + A * u[col]
    gemm_bf16_wmma<<<dim3(Nn / GBN, Nn / GBM, Bb), blk256, 0, stream>>>(
        Acur, NN, Acur, NN, wsU, Nn, 1, wsQ, NN, Nn, Nn, Nn);
    epi_Q<<<ebBNN, blk256, 0, stream>>>(wsQ, Acur, wsU, (long long)BNNf, Nn);
    // R = A @ (u .* Q);  A2 = floor(w[m]*w[l]*(R + u[m]*Q) * 1e4) / 1e4
    gemm_bf16_wmma<<<dim3(Nn / GBN, Nn / GBM, Bb), blk256, 0, stream>>>(
        Acur, NN, wsQ, NN, wsU, Nn, 1, Anext, NN, Nn, Nn, Nn);
    epi_A2<<<ebBNN, blk256, 0, stream>>>(Anext, wsQ, wsU, wsW, (long long)BNNf, Nn);
  };
  coarsen(adj,  wsA1, wsX1,  wsCX1);
  coarsen(wsA1, wsA2, wsCX1, wsCX2);

  // ---- Final dense_gcn(cx2, A_final, W2, b2, mask), fused with mean ----
  rowsum_diag<<<rowGrid, blk256, 0, stream>>>(wsA2, wsRS, wsDG, Nn);
  gcn_scal<<<ebBN, blk256, 0, stream>>>(wsRS, wsDG, wsD, (long long)BNf);
  gemm_bf16_wmma<<<dim3(Ff / GBN, Nn / GBM, Bb), blk256, 0, stream>>>(
      wsCX2, (long long)Nn * Hh, W2, 0, nullptr, 0, 0, wsXW2, (long long)Nn * Ff, Nn, Ff, Hh);
  gemm_bf16_wmma<<<dim3(Ff / GBN, Nn / GBM, Bb), blk256, 0, stream>>>(
      wsA2, NN, wsXW2, (long long)Nn * Ff, wsD, Nn, 1, wsH2, (long long)Nn * Ff, Nn, Ff, Nn);

  // Output: [x1.mean(axis=1) | x2.mean(axis=1)]  -> (B, 384)
  mean_cols<<<dim3(Hh, Bb), blk256, 0, stream>>>(wsX1, out, Nn, Hh, Hh + Ff, 0);
  final_mean<<<dim3(Ff, Bb), blk256, 0, stream>>>(wsH2, wsXW2, wsD, wsDG, b2, mask,
                                                  out, Nn, Ff, Hh + Ff, Hh);
}
